// BrainRegion_15195594293713
// MI455X (gfx1250) — matrix operand order repaired and checked by reading, not compiled, not run on hardware
//
#include <hip/hip_runtime.h>
#include <hip/hip_bf16.h>
#include <math.h>

typedef __attribute__((ext_vector_type(16))) __bf16 v16bf;
typedef __attribute__((ext_vector_type(8)))  __bf16 v8bf;
typedef __attribute__((ext_vector_type(8)))  float  v8f;

constexpr int BSZ  = 8192;   // batch
constexpr int HDIM = 2048;   // hidden / output cols
constexpr int KDIM = 4096;   // folded reduction dim (IN + H)

// ---------------------------------------------------------------------------
// Kernel 1: pack [x_t | state] into bf16 A matrix [BSZ x KDIM], row-major
// ---------------------------------------------------------------------------
__global__ __launch_bounds__(256) void prep_act(const float* __restrict__ x,
                                                const float* __restrict__ st,
                                                __bf16* __restrict__ Abf) {
    size_t idx = (size_t)blockIdx.x * 256 + threadIdx.x;   // < BSZ*KDIM exactly
    int    k   = (int)(idx & (KDIM - 1));
    size_t b   = idx >> 12;                                // / KDIM
    float  v   = (k < HDIM) ? x[b * HDIM + k] : st[b * HDIM + (k - HDIM)];
    Abf[idx] = (__bf16)v;
}

// ---------------------------------------------------------------------------
// Kernel 2: fold U into bottom half of W, convert to bf16, and transpose to
// WT[n, k] (row-major [HDIM x KDIM]) via LDS 32x32 tiles. blockIdx.z: 0=c, 1=g
// ---------------------------------------------------------------------------
__global__ __launch_bounds__(256) void prep_w(const float* __restrict__ Wc,
                                              const float* __restrict__ Uc,
                                              const float* __restrict__ Wg,
                                              const float* __restrict__ Ug,
                                              __bf16* __restrict__ WcT,
                                              __bf16* __restrict__ WgT) {
    __shared__ __bf16 tile[32][33];
    const float* W  = blockIdx.z ? Wg  : Wc;
    const float* U  = blockIdx.z ? Ug  : Uc;
    __bf16*      WT = blockIdx.z ? WgT : WcT;
    int kt = blockIdx.x * 32, nt = blockIdx.y * 32;
    int tx = threadIdx.x, ty = threadIdx.y;         // (32, 8)
#pragma unroll
    for (int i = 0; i < 4; ++i) {                   // coalesced read (n fastest)
        int kl = ty + i * 8, nl = tx;
        int k = kt + kl, n = nt + nl;
        float v = W[(size_t)k * HDIM + n];
        if (k >= HDIM) v += U[(size_t)(k - HDIM) * HDIM + n];
        tile[kl][nl] = (__bf16)v;
    }
    __syncthreads();
#pragma unroll
    for (int i = 0; i < 4; ++i) {                   // coalesced write (k fastest)
        int nl = ty + i * 8, kl = tx;
        WT[(size_t)(nt + nl) * KDIM + kt + kl] = tile[kl][nl];
    }
}

// ---------------------------------------------------------------------------
// Kernel 3: dual GEMM (candidate + gate share the A operand) with bf16 WMMA,
// fused liquid-cell epilogue. 4 waves per WG; wave tile = 32(M) x 64(N) for
// BOTH accumulator sets (16 WMMAs per 20 B128 loads per K-step).
// Writes pre-LayerNorm h_new to d_out.
// ---------------------------------------------------------------------------
__global__ __launch_bounds__(128) void gemm_fused(
    const __bf16* __restrict__ Abf, const __bf16* __restrict__ WcT,
    const __bf16* __restrict__ WgT, const float* __restrict__ bc,
    const float* __restrict__ bg,  const float* __restrict__ logstep,
    const float* __restrict__ state, float* __restrict__ Hout) {
    const int lane  = threadIdx.x & 31;
    const int wave  = threadIdx.x >> 5;
    const int lmod  = lane & 15;        // row (A) or col (B/C) within tile
    const int lhalf = lane >> 4;        // K-half selector per ISA layouts
    const int m0 = blockIdx.y * 128 + wave * 32;   // wave owns 32 rows
    const int n0 = blockIdx.x * 64;

    v8f accC[2][4] = {};
    v8f accG[2][4] = {};

    // A fragments (16-bit A 16x32 layout): lane m = lmod; K-halves split by lhalf.
    const __bf16* aBase0 = Abf + (size_t)(m0 + lmod)      * KDIM + lhalf * 8;
    const __bf16* aBase1 = Abf + (size_t)(m0 + 16 + lmod) * KDIM + lhalf * 8;
    // B fragments (16-bit B 32x16 layout): lane n = lmod; K-halves split by lhalf (same as A).
    const __bf16* bcBase[4];
    const __bf16* bgBase[4];
#pragma unroll
    for (int t = 0; t < 4; ++t) {
        size_t roff = (size_t)(n0 + t * 16 + lmod) * KDIM + lhalf * 8;
        bcBase[t] = WcT + roff;
        bgBase[t] = WgT + roff;
    }

#pragma unroll 2
    for (int kk = 0; kk < KDIM; kk += 32) {
        v16bf a0, a1;
        ((v8bf*)&a0)[0] = *(const v8bf*)(aBase0 + kk);        // K lhalf*8 .. +7
        ((v8bf*)&a0)[1] = *(const v8bf*)(aBase0 + kk + 16);   // K 16+lhalf*8 ..
        ((v8bf*)&a1)[0] = *(const v8bf*)(aBase1 + kk);
        ((v8bf*)&a1)[1] = *(const v8bf*)(aBase1 + kk + 16);
#pragma unroll
        for (int t = 0; t < 4; ++t) {
            v16bf bC;
            ((v8bf*)&bC)[0] = *(const v8bf*)(bcBase[t] + kk);
            ((v8bf*)&bC)[1] = *(const v8bf*)(bcBase[t] + kk + 16);
            accC[0][t] = __builtin_amdgcn_wmma_f32_16x16x32_bf16(
                false, a0, false, bC, (short)0, accC[0][t], false, false);
            accC[1][t] = __builtin_amdgcn_wmma_f32_16x16x32_bf16(
                false, a1, false, bC, (short)0, accC[1][t], false, false);
            v16bf bG;
            ((v8bf*)&bG)[0] = *(const v8bf*)(bgBase[t] + kk);
            ((v8bf*)&bG)[1] = *(const v8bf*)(bgBase[t] + kk + 16);
            accG[0][t] = __builtin_amdgcn_wmma_f32_16x16x32_bf16(
                false, a0, false, bG, (short)0, accG[0][t], false, false);
            accG[1][t] = __builtin_amdgcn_wmma_f32_16x16x32_bf16(
                false, a1, false, bG, (short)0, accG[1][t], false, false);
        }
    }

    // Epilogue. C/D layout: VGPR r -> row base + r + lhalf*8 ; lane -> col n0+t*16+lmod
#pragma unroll
    for (int t = 0; t < 4; ++t) {
        const int n = n0 + t * 16 + lmod;
        const float biasC = bc[n];
        const float biasG = bg[n];
        // tau = exp(log_step); alpha = exp(-1/tau) = exp(-exp(-log_step))
        const float alpha = __expf(-__expf(-logstep[n]));
#pragma unroll
        for (int blk = 0; blk < 2; ++blk) {
#pragma unroll
            for (int r = 0; r < 8; ++r) {
                const int m = m0 + blk * 16 + lhalf * 8 + r;
                float cand = tanhf(accC[blk][t][r] + biasC);
                float gate = 1.0f / (1.0f + __expf(-(accG[blk][t][r] + biasG)));
                float s    = state[(size_t)m * HDIM + n];
                Hout[(size_t)m * HDIM + n] =
                    alpha * s + (1.0f - alpha) * gate * cand;
            }
        }
    }
}

// ---------------------------------------------------------------------------
// Kernel 4: in-place row LayerNorm of d_out (one block per row of 2048).
// ---------------------------------------------------------------------------
__global__ __launch_bounds__(256) void layernorm_k(float* __restrict__ out,
                                                   const float* __restrict__ gamma,
                                                   const float* __restrict__ beta) {
    __shared__ float s1[256];
    __shared__ float s2[256];
    const int row = blockIdx.x, tid = threadIdx.x;
    float* xr = out + (size_t)row * HDIM;
    float vals[8];
    float sum = 0.f, sq = 0.f;
#pragma unroll
    for (int j = 0; j < 8; ++j) {
        float v = xr[tid + j * 256];
        vals[j] = v;
        sum += v;
        sq  += v * v;
    }
    s1[tid] = sum; s2[tid] = sq;
    __syncthreads();
    for (int off = 128; off > 0; off >>= 1) {
        if (tid < off) { s1[tid] += s1[tid + off]; s2[tid] += s2[tid + off]; }
        __syncthreads();
    }
    const float mu  = s1[0] * (1.0f / HDIM);
    const float var = s2[0] * (1.0f / HDIM) - mu * mu;
    const float rs  = rsqrtf(var + 1e-5f);
#pragma unroll
    for (int j = 0; j < 8; ++j) {
        int c = tid + j * 256;
        xr[c] = (vals[j] - mu) * rs * gamma[c] + beta[c];
    }
}

// ---------------------------------------------------------------------------
extern "C" void kernel_launch(void* const* d_in, const int* in_sizes, int n_in,
                              void* d_out, int out_size, void* d_ws, size_t ws_size,
                              hipStream_t stream) {
    const float* x_t      = (const float*)d_in[0];
    const float* state    = (const float*)d_in[1];
    const float* Wc       = (const float*)d_in[2];
    const float* Uc       = (const float*)d_in[3];
    const float* bc       = (const float*)d_in[4];
    const float* Wg       = (const float*)d_in[5];
    const float* Ug       = (const float*)d_in[6];
    const float* bg       = (const float*)d_in[7];
    const float* log_step = (const float*)d_in[8];
    const float* gamma    = (const float*)d_in[9];
    const float* beta     = (const float*)d_in[10];
    float* out = (float*)d_out;

    // Workspace layout (96 MiB total)
    char* ws = (char*)d_ws;
    constexpr size_t ABF_BYTES = (size_t)BSZ * KDIM * 2;   // 64 MiB
    constexpr size_t WT_BYTES  = (size_t)HDIM * KDIM * 2;  // 16 MiB each
    __bf16* Abf = (__bf16*)ws;
    __bf16* WcT = (__bf16*)(ws + ABF_BYTES);
    __bf16* WgT = (__bf16*)(ws + ABF_BYTES + WT_BYTES);

    // 1) pack activations to bf16
    prep_act<<<(BSZ * (size_t)KDIM) / 256, 256, 0, stream>>>(x_t, state, Abf);

    // 2) fold U into W, transpose, convert to bf16
    prep_w<<<dim3(KDIM / 32, HDIM / 32, 2), dim3(32, 8), 0, stream>>>(
        Wc, Uc, Wg, Ug, WcT, WgT);

    // 3) dual WMMA GEMM + liquid-cell epilogue -> pre-norm h in d_out
    gemm_fused<<<dim3(HDIM / 64, BSZ / 128), 128, 0, stream>>>(
        Abf, WcT, WgT, bc, bg, log_step, state, out);

    // 4) in-place LayerNorm
    layernorm_k<<<BSZ, 256, 0, stream>>>(out, gamma, beta);
}
